// GCNLayer_64450279243778
// MI455X (gfx1250) — matrix operand order repaired and checked by reading, not compiled, not run on hardware
//
#include <hip/hip_runtime.h>
#include <hip/hip_bf16.h>

// ---------------------------------------------------------------------------
// GCN layer on gfx1250 (MI455X):
//   T1    = A @ X + X                      (A2 = A + I applied implicitly)
//   temp2 = dscale .* (T1 @ W^T)           (dscale = (diag(A)+1)^-1/2)
//   out   = relu(dscale .* (A @ temp2 + temp2))
// GEMMs: bf16 split-precision (hi+lo), 3x v_wmma_f32_16x16x32_bf16 per tile
// (hh + hl + lh), fp32 accumulate -> fp32-class accuracy.
// B operands are pre-split + transposed into packed bf16 [N][K]; per the
// k order this code uses (lane n=L%16, K consecutive per lane half, as for A) a
// B fragment is one 32-byte contiguous load per lane, zero hot-loop VALU.
// ---------------------------------------------------------------------------

typedef __bf16 bf16_t;
typedef bf16_t v16bf __attribute__((ext_vector_type(16)));
typedef float  v8f   __attribute__((ext_vector_type(8)));

#define GCN_N   4096
#define GCN_D   512
#define WN      4          // 16-wide N-tiles per wave -> 16x64 strip per wave

__device__ __forceinline__ void split_bf16(float a, bf16_t& h, bf16_t& l) {
    h = (bf16_t)a;               // RNE to bf16
    l = (bf16_t)(a - (float)h);  // residual (exact subtract, then RNE)
}

// A fragment (16x32, row-major fp32 source), contiguous k order (same as B):
// lanes 0-15: K = kb..kb+15; lanes 16-31: K = kb+16..kb+31.
// base = row_ptr + kb + (lane>>4)*16.
__device__ __forceinline__ void load_fragA(const float* __restrict__ base,
                                           v16bf& hi, v16bf& lo) {
    const float4* p4 = (const float4*)base;
    float4 q0 = p4[0], q1 = p4[1];   // K = +0 .. +7
    float4 q2 = p4[2], q3 = p4[3];   // K = +8 .. +15
    float v[16] = {q0.x,q0.y,q0.z,q0.w, q1.x,q1.y,q1.z,q1.w,
                   q2.x,q2.y,q2.z,q2.w, q3.x,q3.y,q3.z,q3.w};
#pragma unroll
    for (int j = 0; j < 16; ++j) {
        bf16_t h, l; split_bf16(v[j], h, l);
        hi[j] = h; lo[j] = l;
    }
}

// B fragment (32x16) from pre-packed bf16 [N][K], same contiguous k order as A:
// lane n = L%16; lanes 0-15 hold K = kb..kb+15, lanes 16-31 hold kb+16..kb+31
// -> 16 consecutive bf16 per lane = one 32-byte aligned vector load.
__device__ __forceinline__ v16bf load_fragB(const bf16_t* __restrict__ base) {
    return *(const v16bf*)base;
}

// MODE 0: epilogue C += addRow[row,col]               (T1 = A@X + X)
// MODE 1: epilogue C *= dscale[row]                   (temp2)
// MODE 2: epilogue relu(dscale[row]*(C + addRow))     (out)
template <int MODE>
__global__ __launch_bounds__(256)
void gcn_gemm_bf16x3(const float* __restrict__ Aop, int lda,
                     const bf16_t* __restrict__ Bh,
                     const bf16_t* __restrict__ Bl,   // packed [Ncols][K]
                     float* __restrict__ Cout, int ldc,
                     int Ncols, int K,
                     const float* __restrict__ dscale,
                     const float* __restrict__ addRow) {
    const int lane   = threadIdx.x & 31;
    const int wid    = blockIdx.x * (blockDim.x >> 5) + (threadIdx.x >> 5);
    const int tilesN = Ncols / (16 * WN);         // 8 for D=512
    const int tm     = (wid / tilesN) * 16;
    const int tn0    = (wid % tilesN) * (16 * WN);

    const int lhalf  = lane >> 4;                 // lane half
    const int lmod   = lane & 15;

    v8f acc[WN] = {};

    const float* arow_ptr = Aop + (size_t)(tm + lmod) * lda;
    const int    bkoff    = lhalf * 16;           // B: consecutive K per half

    for (int kb = 0; kb < K; kb += 32) {
        __builtin_prefetch(arow_ptr + kb + lhalf * 8 + 32, 0, 0); // next A blk

        v16bf ahi, alo;
        load_fragA(arow_ptr + kb + lhalf * 16, ahi, alo);

#pragma unroll
        for (int t = 0; t < WN; ++t) {
            const size_t boff = (size_t)(tn0 + t * 16 + lmod) * K + kb + bkoff;
            v16bf bhi = load_fragB(Bh + boff);
            v16bf blo = load_fragB(Bl + boff);
            acc[t] = __builtin_amdgcn_wmma_f32_16x16x32_bf16(
                         false, alo, false, bhi, (short)0, acc[t], false, false);
            acc[t] = __builtin_amdgcn_wmma_f32_16x16x32_bf16(
                         false, ahi, false, blo, (short)0, acc[t], false, false);
            acc[t] = __builtin_amdgcn_wmma_f32_16x16x32_bf16(
                         false, ahi, false, bhi, (short)0, acc[t], false, false);
        }
    }

    // Epilogue. C/D layout: VGPR r -> M = tm + r + 8*lhalf, N = tn + lmod.
#pragma unroll
    for (int t = 0; t < WN; ++t) {
        const int col = tn0 + t * 16 + lmod;
#pragma unroll
        for (int r = 0; r < 8; ++r) {
            const int row = tm + r + lhalf * 8;
            float v = acc[t][r];
            if constexpr (MODE == 0 || MODE == 2)
                v += addRow[(size_t)row * ldc + col];     // the "+ I @ B" term
            if constexpr (MODE == 1 || MODE == 2)
                v *= dscale[row];
            if constexpr (MODE == 2)
                v = fmaxf(v, 0.0f);
            Cout[(size_t)row * ldc + col] = v;
        }
    }
}

// Transpose + split fp32 [K][N] -> bf16 hi/lo [N][K]; LDS-tiled for coalescing.
__global__ __launch_bounds__(256)
void gcn_pack_bt(const float* __restrict__ B, int K, int N,
                 bf16_t* __restrict__ bh, bf16_t* __restrict__ bl) {
    __shared__ float tile[32][33];
    const int k0 = blockIdx.x * 32;
    const int n0 = blockIdx.y * 32;
    const int tx = threadIdx.x;        // 0..31
    const int ty = threadIdx.y;        // 0..7
#pragma unroll
    for (int i = 0; i < 32; i += 8)
        tile[ty + i][tx] = B[(size_t)(k0 + ty + i) * N + (n0 + tx)];
    __syncthreads();
#pragma unroll
    for (int i = 0; i < 32; i += 8) {
        const int n = n0 + ty + i;
        const int k = k0 + tx;
        float a = tile[tx][ty + i];
        bf16_t h, l; split_bf16(a, h, l);
        bh[(size_t)n * K + k] = h;
        bl[(size_t)n * K + k] = l;
    }
}

// Elementwise split (no transpose) for W, already stored [N][K].
__global__ void gcn_pack_rows(const float* __restrict__ src,
                              bf16_t* __restrict__ bh, bf16_t* __restrict__ bl,
                              int count) {
    int i = blockIdx.x * blockDim.x + threadIdx.x;
    if (i < count) {
        bf16_t h, l; split_bf16(src[i], h, l);
        bh[i] = h; bl[i] = l;
    }
}

__global__ void gcn_dscale(const float* __restrict__ A, float* __restrict__ d,
                           int n, int lda) {
    int i = blockIdx.x * blockDim.x + threadIdx.x;
    if (i < n) d[i] = 1.0f / sqrtf(A[(size_t)i * lda + i] + 1.0f);
}

extern "C" void kernel_launch(void* const* d_in, const int* in_sizes, int n_in,
                              void* d_out, int out_size, void* d_ws, size_t ws_size,
                              hipStream_t stream) {
    const float* X = (const float*)d_in[0];   // [N, D]
    const float* A = (const float*)d_in[1];   // [N, N]
    const float* W = (const float*)d_in[2];   // [D, D] (out, in)
    float* out = (float*)d_out;               // [N, D]

    const int N = GCN_N, D = GCN_D;

    float*  T1    = (float*)d_ws;                    // [N, D]   8 MB
    float*  temp2 = T1 + (size_t)N * D;              // [N, D]   8 MB
    float*  dsc   = temp2 + (size_t)N * D;           // [N]      16 KB
    bf16_t* Bth   = (bf16_t*)(dsc + N);              // [D, N]   4 MB (reused)
    bf16_t* Btl   = Bth + (size_t)D * N;             // [D, N]   4 MB
    bf16_t* Wh    = Btl + (size_t)D * N;             // [D, D]   0.5 MB
    bf16_t* Wl    = Wh + (size_t)D * D;              // [D, D]   0.5 MB

    // dscale = (diag(A) + 1)^-1/2
    gcn_dscale<<<(N + 255) / 256, 256, 0, stream>>>(A, dsc, N, N);

    dim3 tblk(32, 8);
    dim3 tg1(N / 32, D / 32);                        // transpose X / temp2

    // 16x64 strip per wave, 8 waves (256 threads) per block
    const int tilesN = D / (16 * WN);                // 8
    const int waves  = (N / 16) * tilesN;            // 2048
    dim3 blk(256), grd(waves / 8);                   // 256 blocks

    // --- GEMM1: T1 = A @ X + X ---
    gcn_pack_bt<<<tg1, tblk, 0, stream>>>(X, N, D, Bth, Btl);
    gcn_gemm_bf16x3<0><<<grd, blk, 0, stream>>>(A, N, Bth, Btl, T1, D, D, N,
                                                nullptr, X);
    // --- GEMM2: temp2 = dscale .* (T1 @ W^T) ---
    gcn_pack_rows<<<(D * D + 255) / 256, 256, 0, stream>>>(W, Wh, Wl, D * D);
    gcn_gemm_bf16x3<1><<<grd, blk, 0, stream>>>(T1, D, Wh, Wl, temp2, D, D, D,
                                                dsc, nullptr);
    // --- GEMM3: out = relu(dscale .* (A @ temp2 + temp2)) ---
    gcn_pack_bt<<<tg1, tblk, 0, stream>>>(temp2, N, D, Bth, Btl);
    gcn_gemm_bf16x3<2><<<grd, blk, 0, stream>>>(A, N, Bth, Btl, out, D, D, N,
                                                dsc, temp2);
}
